// MemTransformerLM_72172630442407
// MI455X (gfx1250) — matrix operand order repaired and checked by reading, not compiled, not run on hardware
//
#include <hip/hip_runtime.h>
#include <hip/hip_bf16.h>

// ---------------------------------------------------------------------------
// MemTransformerLM relative-attention layer for gfx1250 (MI455X), bf16 WMMA,
// software-pipelined GEMM tiles + async global->LDS staging.
// ---------------------------------------------------------------------------

typedef __attribute__((ext_vector_type(16))) __bf16 v16bf;
typedef __attribute__((ext_vector_type(8)))  __bf16 v8bf;
typedef __attribute__((ext_vector_type(8)))  float  v8f;

#define QLEN 512
#define MLEN 512
#define KLEN 1024
#define BSZ  8
#define NH   16
#define DH   64
#define DM   1024

// --- WMMA fragment helpers (A/B in contiguous k order k = 16*(lane/16) + s, wave32) ---------------------
// A (16x32 bf16): lane = M (0..15), lanes16-31 same M with K offset 16.
// VGPR 0..3 hold K = kb+0..7 ; VGPR 4..7 hold K = kb+8..15.
static __device__ __forceinline__ v16bf load_a_frag(const __bf16* A, int lda, int lane) {
  const int m  = lane & 15;
  const int kb = (lane >> 4) << 4;            // 0 or 16
  const __bf16* row = A + m * lda + kb;
  v8bf lo = *(const v8bf*)(row);              // K = kb .. kb+7
  v8bf hi = *(const v8bf*)(row + 8);          // K = kb+8 .. kb+15
  v16bf a;
#pragma unroll
  for (int i = 0; i < 8; ++i) { a[i] = lo[i]; a[8 + i] = hi[i]; }
  return a;
}

// B (32x16 bf16), supplied as Bt = N x K row-major (row n contiguous in K):
// lane = N, lanes 0-15 hold K=0..15, lanes 16-31 hold K=16..31.
static __device__ __forceinline__ v16bf load_b_frag(const __bf16* Bt, int ldb, int lane) {
  return *(const v16bf*)(Bt + (size_t)(lane & 15) * ldb + ((lane >> 4) << 4));
}

#define WMMA(a, b, c) \
  __builtin_amdgcn_wmma_f32_16x16x32_bf16(false, (a), false, (b), (short)0, (c), false, false)

// C/D tile element (v, lane) -> M = v + 8*(lane>>4), N = lane&15.
#define C_ROW(v, lane) ((v) + (((lane) >> 4) << 3))
#define C_COL(lane)    ((lane) & 15)

// Software-pipelined 16(M) x 64(N) x K tile: fragments for step kk+32 are
// issued before the WMMAs of step kk, so VMEM overlaps the matrix pipe.
static __device__ __forceinline__ void gemm_tile_16x64(
    const __bf16* __restrict__ A, int lda,
    const __bf16* __restrict__ Bt, int ldb,
    int K, int lane, v8f acc[4]) {
  v16bf a  = load_a_frag(A, lda, lane);
  v16bf b0 = load_b_frag(Bt + (size_t)0  * ldb, ldb, lane);
  v16bf b1 = load_b_frag(Bt + (size_t)16 * ldb, ldb, lane);
  v16bf b2 = load_b_frag(Bt + (size_t)32 * ldb, ldb, lane);
  v16bf b3 = load_b_frag(Bt + (size_t)48 * ldb, ldb, lane);
  for (int kk = 32; kk < K; kk += 32) {
    v16bf an = load_a_frag(A + kk, lda, lane);
    v16bf c0 = load_b_frag(Bt + (size_t)0  * ldb + kk, ldb, lane);
    v16bf c1 = load_b_frag(Bt + (size_t)16 * ldb + kk, ldb, lane);
    v16bf c2 = load_b_frag(Bt + (size_t)32 * ldb + kk, ldb, lane);
    v16bf c3 = load_b_frag(Bt + (size_t)48 * ldb + kk, ldb, lane);
    acc[0] = WMMA(a, b0, acc[0]);
    acc[1] = WMMA(a, b1, acc[1]);
    acc[2] = WMMA(a, b2, acc[2]);
    acc[3] = WMMA(a, b3, acc[3]);
    a = an; b0 = c0; b1 = c1; b2 = c2; b3 = c3;
  }
  acc[0] = WMMA(a, b0, acc[0]);
  acc[1] = WMMA(a, b1, acc[1]);
  acc[2] = WMMA(a, b2, acc[2]);
  acc[3] = WMMA(a, b3, acc[3]);
}

// --- 0: fp32 -> bf16 convert ------------------------------------------------
__global__ __launch_bounds__(256) void k_cvt(const float* __restrict__ src,
                                             __bf16* __restrict__ dst, int n) {
  int i = blockIdx.x * 256 + threadIdx.x;
  if (i < n) dst[i] = (__bf16)src[i];
}

// --- 1: QKV projection: cat(8192x1024) @ Wqkv^T(1024x3072) ------------------
// Epilogue scatters into qw/qr (with biases), k[b,n,j,d], vT[b,n,d,j].
__global__ __launch_bounds__(256)
void k_qkv(const __bf16* __restrict__ A, const __bf16* __restrict__ Bt,
           const float* __restrict__ rwb, const float* __restrict__ rrb,
           __bf16* __restrict__ qw, __bf16* __restrict__ qr,
           __bf16* __restrict__ kb, __bf16* __restrict__ vT) {
  const int lane = threadIdx.x & 31, wave = threadIdx.x >> 5;
  const int tile = blockIdx.x * 8 + wave;          // 512 x 48 tiles
  const int mt = tile / 48, nt = tile % 48;
  const int row0 = mt * 16, col0 = nt * 64;
  v8f acc[4] = {};
  gemm_tile_16x64(A + (size_t)row0 * DM, DM, Bt + (size_t)col0 * DM, DM, DM, lane, acc);
#pragma unroll
  for (int t = 0; t < 4; ++t)
#pragma unroll
    for (int v = 0; v < 8; ++v) {
      const int gr = row0 + C_ROW(v, lane);        // global row = j*BSZ + b
      const int gc = col0 + 16 * t + C_COL(lane);  // global col in [0,3072)
      const int j = gr >> 3, bb = gr & 7;
      const int seg = gc >> 10, rem = gc & 1023, nh = rem >> 6, d = rem & 63;
      const float val = acc[t][v];
      if (seg == 0) {
        if (j >= MLEN) {
          const int i = j - MLEN;
          const size_t idx = (((size_t)(bb * NH + nh) * QLEN) + i) * DH + d;
          qw[idx] = (__bf16)(val + rwb[nh * DH + d]);
          qr[idx] = (__bf16)(val + rrb[nh * DH + d]);
        }
      } else if (seg == 1) {
        kb[(((size_t)(bb * NH + nh) * KLEN) + j) * DH + d] = (__bf16)val;
      } else {
        vT[(((size_t)(bb * NH + nh) * DH) + d) * KLEN + j] = (__bf16)val;
      }
    }
}

// --- 2: r_head_k: r(1024x1024) @ Wr^T -> rk[n][j][d] ------------------------
__global__ __launch_bounds__(256)
void k_rk(const __bf16* __restrict__ A, const __bf16* __restrict__ Bt,
          __bf16* __restrict__ rk) {
  const int lane = threadIdx.x & 31, wave = threadIdx.x >> 5;
  const int tile = blockIdx.x * 8 + wave;          // 64 x 16 tiles
  const int mt = tile >> 4, nt = tile & 15;
  const int row0 = mt * 16, col0 = nt * 64;
  v8f acc[4] = {};
  gemm_tile_16x64(A + (size_t)row0 * DM, DM, Bt + (size_t)col0 * DM, DM, DM, lane, acc);
#pragma unroll
  for (int t = 0; t < 4; ++t)
#pragma unroll
    for (int v = 0; v < 8; ++v) {
      const int j = row0 + C_ROW(v, lane);
      const int gc = col0 + 16 * t + C_COL(lane);
      const int nh = gc >> 6, d = gc & 63;
      rk[(((size_t)nh * KLEN) + j) * DH + d] = (__bf16)acc[t][v];
    }
}

// --- 3: BD_raw[b,n] = qr[b,n](512x64) @ rk[n]^T -----------------------------
__global__ __launch_bounds__(256)
void k_bd(const __bf16* __restrict__ qr, const __bf16* __restrict__ rk,
          float* __restrict__ bd) {
  const int lane = threadIdx.x & 31, wave = threadIdx.x >> 5;
  const int tile = blockIdx.x * 8 + wave;          // 128 bn x 32 it x 16 jt
  const int jt = tile & 15, it = (tile >> 4) & 31, bn = tile >> 9;
  const int n = bn & 15;
  const __bf16* Aq = qr + (size_t)bn * QLEN * DH + (size_t)(it * 16) * DH;
  const __bf16* Bt = rk + (size_t)n * KLEN * DH + (size_t)(jt * 64) * DH;
  v8f acc[4] = {};
  gemm_tile_16x64(Aq, DH, Bt, DH, DH, lane, acc);
  float* dst = bd + ((size_t)bn * QLEN + it * 16) * KLEN + jt * 64;
#pragma unroll
  for (int t = 0; t < 4; ++t)
#pragma unroll
    for (int v = 0; v < 8; ++v)
      dst[(size_t)C_ROW(v, lane) * KLEN + 16 * t + C_COL(lane)] = acc[t][v];
}

// --- 4: AC GEMM + rel_shift + mask + scale, in place over BD buffer ---------
// One block per (b,n,i-tile of 16 rows). BD rows are staged to LDS with
// GLOBAL_LOAD_ASYNC_TO_LDS_B128 (ASYNCcnt path, no VGPR round-trip), then
// score = (AC[i,j] + BD[i, j-i+511]) * scale for j<=i+512 else -1e30,
// overwriting the BD buffer.
__global__ __launch_bounds__(256)
void k_ac(const __bf16* __restrict__ qw, const __bf16* __restrict__ kmat,
          float* __restrict__ sc) {
  __shared__ float lbd[16 * KLEN];                 // 64 KB
  const int bn = blockIdx.x >> 5, it = blockIdx.x & 31;
  const int i0 = it * 16;
  float* rowblk = sc + ((size_t)bn * QLEN + i0) * KLEN;

  // Async-copy 16x1024 f32 (64 KB) global -> LDS: 4096 x B128, 16 per thread.
  {
    const unsigned lds_base = (unsigned)(uintptr_t)&lbd[0];
    const char* gbase = (const char*)rowblk;
#pragma unroll
    for (int c = 0; c < 16; ++c) {
      const unsigned boff = (threadIdx.x + c * 256) * 16u;
      const unsigned ldsa = lds_base + boff;
      const unsigned long long ga = (unsigned long long)(uintptr_t)(gbase + boff);
      asm volatile("global_load_async_to_lds_b128 %0, %1, off"
                   :: "v"(ldsa), "v"(ga) : "memory");
    }
    asm volatile("s_wait_asynccnt 0x0" ::: "memory");
  }
  __syncthreads();

  const int lane = threadIdx.x & 31, wave = threadIdx.x >> 5;
  const __bf16* Aq = qw + (size_t)bn * QLEN * DH + (size_t)i0 * DH;
  const __bf16* Bt = kmat + (size_t)bn * KLEN * DH;
  const int col0 = wave * 128;
  v16bf a0 = load_a_frag(Aq, DH, lane);
  v16bf a1 = load_a_frag(Aq + 32, DH, lane);
  v8f acc[8] = {};
  // double-buffered B fragments across the 8 sub-tiles
  v16bf b0 = load_b_frag(Bt + (size_t)col0 * DH, DH, lane);
  v16bf b1 = load_b_frag(Bt + (size_t)col0 * DH + 32, DH, lane);
#pragma unroll
  for (int t = 0; t < 8; ++t) {
    const int tn = (t < 7) ? (t + 1) : 7;
    const __bf16* bp = Bt + (size_t)(col0 + 16 * tn) * DH;
    v16bf n0 = load_b_frag(bp, DH, lane);
    v16bf n1 = load_b_frag(bp + 32, DH, lane);
    acc[t] = WMMA(a0, b0, acc[t]);
    acc[t] = WMMA(a1, b1, acc[t]);
    b0 = n0; b1 = n1;
  }
  const float scale = 0.125f;                      // 1/sqrt(64)
#pragma unroll
  for (int t = 0; t < 8; ++t)
#pragma unroll
    for (int v = 0; v < 8; ++v) {
      const int m = C_ROW(v, lane);
      const int i = i0 + m;
      const int j = col0 + 16 * t + C_COL(lane);
      float s;
      if (j > i + MLEN) {
        s = -1e30f;
      } else {
        const int sidx = j - i + (QLEN - 1);       // in [0, KLEN-1]
        s = (acc[t][v] + lbd[m * KLEN + sidx]) * scale;
      }
      rowblk[(size_t)m * KLEN + j] = s;
    }
}

// --- 5: row softmax -> bf16 probs -------------------------------------------
__global__ __launch_bounds__(256)
void k_softmax(const float* __restrict__ sc, __bf16* __restrict__ prob) {
  const int row = blockIdx.x * 8 + (threadIdx.x >> 5);
  const int lane = threadIdx.x & 31;
  const float* s = sc + (size_t)row * KLEN;
  float vals[32];
  float mx = -1e30f;
#pragma unroll
  for (int t = 0; t < 32; ++t) { vals[t] = s[lane + 32 * t]; mx = fmaxf(mx, vals[t]); }
  for (int off = 16; off; off >>= 1) mx = fmaxf(mx, __shfl_xor(mx, off, 32));
  float sum = 0.f;
#pragma unroll
  for (int t = 0; t < 32; ++t) { vals[t] = __expf(vals[t] - mx); sum += vals[t]; }
  for (int off = 16; off; off >>= 1) sum += __shfl_xor(sum, off, 32);
  const float inv = 1.f / sum;
  __bf16* p = prob + (size_t)row * KLEN;
#pragma unroll
  for (int t = 0; t < 32; ++t) p[lane + 32 * t] = (__bf16)(vals[t] * inv);
}

// --- 6: attn_vec[b,n] = prob(512x1024) @ v(1024x64), vT supplies Bt ---------
__global__ __launch_bounds__(256)
void k_pv(const __bf16* __restrict__ prob, const __bf16* __restrict__ vT,
          __bf16* __restrict__ avec) {
  const int lane = threadIdx.x & 31, wave = threadIdx.x >> 5;
  const int tile = blockIdx.x * 8 + wave;          // 128 bn x 32 it
  const int it = tile & 31, bn = tile >> 5;
  const int b = bn >> 4, n = bn & 15;
  const __bf16* A  = prob + (size_t)bn * QLEN * KLEN + (size_t)(it * 16) * KLEN;
  const __bf16* Bt = vT + (size_t)bn * DH * KLEN;
  v8f acc[4] = {};
  gemm_tile_16x64(A, KLEN, Bt, KLEN, KLEN, lane, acc);
#pragma unroll
  for (int t = 0; t < 4; ++t)
#pragma unroll
    for (int v = 0; v < 8; ++v) {
      const int i = it * 16 + C_ROW(v, lane);
      const int d = 16 * t + C_COL(lane);
      avec[((size_t)(i * BSZ + b)) * DM + n * DH + d] = (__bf16)acc[t][v];
    }
}

// --- 7: attn_out = attn_vec(4096x1024) @ Wo^T -------------------------------
__global__ __launch_bounds__(256)
void k_out(const __bf16* __restrict__ A, const __bf16* __restrict__ Bt,
           float* __restrict__ out) {
  const int lane = threadIdx.x & 31, wave = threadIdx.x >> 5;
  const int tile = blockIdx.x * 8 + wave;          // 256 mt x 16 nt
  const int mt = tile >> 4, nt = tile & 15;
  const int row0 = mt * 16, col0 = nt * 64;
  v8f acc[4] = {};
  gemm_tile_16x64(A + (size_t)row0 * DM, DM, Bt + (size_t)col0 * DM, DM, DM, lane, acc);
#pragma unroll
  for (int t = 0; t < 4; ++t)
#pragma unroll
    for (int v = 0; v < 8; ++v)
      out[(size_t)(row0 + C_ROW(v, lane)) * DM + col0 + 16 * t + C_COL(lane)] = acc[t][v];
}

// --- 8: LayerNorm(w + attn_out) ---------------------------------------------
__global__ __launch_bounds__(256)
void k_ln(const float* __restrict__ w, const float* __restrict__ ao,
          const float* __restrict__ g, const float* __restrict__ bta,
          float* __restrict__ out) {
  const int row = blockIdx.x * 8 + (threadIdx.x >> 5);
  const int lane = threadIdx.x & 31;
  const float* x1 = w + (size_t)row * DM;
  const float* x2 = ao + (size_t)row * DM;
  float x[32];
  float s = 0.f;
#pragma unroll
  for (int t = 0; t < 32; ++t) { x[t] = x1[lane + 32 * t] + x2[lane + 32 * t]; s += x[t]; }
  for (int off = 16; off; off >>= 1) s += __shfl_xor(s, off, 32);
  const float mu = s * (1.0f / DM);
  float vs = 0.f;
#pragma unroll
  for (int t = 0; t < 32; ++t) { const float d = x[t] - mu; vs += d * d; }
  for (int off = 16; off; off >>= 1) vs += __shfl_xor(vs, off, 32);
  const float rstd = rsqrtf(vs * (1.0f / DM) + 1e-5f);
  float* o = out + (size_t)row * DM;
#pragma unroll
  for (int t = 0; t < 32; ++t) {
    const int c = lane + 32 * t;
    o[c] = (x[t] - mu) * rstd * g[c] + bta[c];
  }
}

// ---------------------------------------------------------------------------
extern "C" void kernel_launch(void* const* d_in, const int* in_sizes, int n_in,
                              void* d_out, int out_size, void* d_ws, size_t ws_size,
                              hipStream_t stream) {
  (void)in_sizes; (void)n_in; (void)out_size; (void)ws_size;
  const float* w    = (const float*)d_in[0];
  const float* mems = (const float*)d_in[1];
  const float* r    = (const float*)d_in[2];
  const float* Wqkv = (const float*)d_in[3];
  const float* Wr   = (const float*)d_in[4];
  const float* Wo   = (const float*)d_in[5];
  const float* rwb  = (const float*)d_in[6];
  const float* rrb  = (const float*)d_in[7];
  const float* lng  = (const float*)d_in[8];
  const float* lnb  = (const float*)d_in[9];
  float* out = (float*)d_out;

  char* ws = (char*)d_ws;
  size_t off = 0;
  auto alloc = [&](size_t bytes) -> void* {
    void* p = ws + off;
    off = (off + bytes + 255) & ~(size_t)255;
    return p;
  };
  __bf16* cat_bf   = (__bf16*)alloc((size_t)KLEN * BSZ * DM * 2);          // 16 MB
  __bf16* Wqkv_bf  = (__bf16*)alloc((size_t)3 * DM * DM * 2);              // 6 MB
  __bf16* Wr_bf    = (__bf16*)alloc((size_t)DM * DM * 2);                  // 2 MB
  __bf16* Wo_bf    = (__bf16*)alloc((size_t)DM * DM * 2);                  // 2 MB
  __bf16* r_bf     = (__bf16*)alloc((size_t)KLEN * DM * 2);                // 2 MB
  __bf16* qw_bf    = (__bf16*)alloc((size_t)BSZ * NH * QLEN * DH * 2);     // 8 MB
  __bf16* qr_bf    = (__bf16*)alloc((size_t)BSZ * NH * QLEN * DH * 2);     // 8 MB
  __bf16* k_bf     = (__bf16*)alloc((size_t)BSZ * NH * KLEN * DH * 2);     // 16 MB
  __bf16* vT_bf    = (__bf16*)alloc((size_t)BSZ * NH * DH * KLEN * 2);     // 16 MB
  __bf16* rk_bf    = (__bf16*)alloc((size_t)NH * KLEN * DH * 2);           // 2 MB
  float*  score    = (float*)alloc((size_t)BSZ * NH * QLEN * KLEN * 4);    // 256 MB (BD then score)
  __bf16* prob_bf  = (__bf16*)alloc((size_t)BSZ * NH * QLEN * KLEN * 2);   // 128 MB
  __bf16* avec_bf  = (__bf16*)alloc((size_t)QLEN * BSZ * DM * 2);          // 8 MB
  float*  aout     = (float*)alloc((size_t)QLEN * BSZ * DM * 4);           // 16 MB

  // 0. converts (cat = [mems; w] is contiguous in row-major (klen,bsz,dm))
  const int nhalf = MLEN * BSZ * DM;
  k_cvt<<<(nhalf + 255) / 256, 256, 0, stream>>>(mems, cat_bf, nhalf);
  k_cvt<<<(nhalf + 255) / 256, 256, 0, stream>>>(w, cat_bf + nhalf, nhalf);
  k_cvt<<<(3 * DM * DM + 255) / 256, 256, 0, stream>>>(Wqkv, Wqkv_bf, 3 * DM * DM);
  k_cvt<<<(DM * DM + 255) / 256, 256, 0, stream>>>(Wr, Wr_bf, DM * DM);
  k_cvt<<<(DM * DM + 255) / 256, 256, 0, stream>>>(Wo, Wo_bf, DM * DM);
  k_cvt<<<(KLEN * DM + 255) / 256, 256, 0, stream>>>(r, r_bf, KLEN * DM);

  // 1. QKV projection (8192x3072x1024) : 512*48 wave tiles / 8 waves
  k_qkv<<<3072, 256, 0, stream>>>(cat_bf, Wqkv_bf, rwb, rrb, qw_bf, qr_bf, k_bf, vT_bf);
  // 2. r_head_k (1024x1024x1024): 64*16 tiles
  k_rk<<<128, 256, 0, stream>>>(r_bf, Wr_bf, rk_bf);
  // 3. BD_raw: 128 * 32 * 16 tiles
  k_bd<<<8192, 256, 0, stream>>>(qr_bf, rk_bf, score);
  // 4. AC + rel_shift + mask + scale (in place): 128 bn * 32 i-tiles blocks
  k_ac<<<4096, 256, 0, stream>>>(qw_bf, k_bf, score);
  // 5. softmax: 128*512 rows / 8 waves
  k_softmax<<<8192, 256, 0, stream>>>(score, prob_bf);
  // 6. PV: 128*32 tiles
  k_pv<<<512, 256, 0, stream>>>(prob_bf, vT_bf, avec_bf);
  // 7. output projection: 256*16 tiles
  k_out<<<512, 256, 0, stream>>>(avec_bf, Wo_bf, aout);
  // 8. residual + layernorm: 4096 rows
  k_ln<<<512, 256, 0, stream>>>(w, aout, lng, lnb, out);
}
